// FeedForwardECMoe_2233382994610
// MI455X (gfx1250) — compile-verified, matrix operand order checked by reading, not run on hardware
//
#include <hip/hip_runtime.h>
#include <cstdint>
#include <cstddef>

// ---------------------------------------------------------------------------
// Expert-choice MoE FFN for MI455X (gfx1250, wave32, WMMA).
// Compute-bound (~0.55 TFLOP GEMM) -> bf16 WMMA (v_wmma_f32_16x16x32_bf16),
// f32 accumulate. Weights pre-transposed so both GEMM operands stage into LDS
// as straight 16B copies via GLOBAL_LOAD_ASYNC_TO_LDS_B128 (ASYNCcnt).
// Deterministic combine (no float atomics).
// ---------------------------------------------------------------------------

typedef __attribute__((ext_vector_type(16))) __bf16 v16bf;
typedef __attribute__((ext_vector_type(8)))  __bf16 v8bf;
typedef __attribute__((ext_vector_type(8)))  float  v8f;
typedef int vi4 __attribute__((__vector_size__(16)));

namespace {
constexpr int kB = 4;      // batch
constexpr int kT = 4096;   // tokens
constexpr int kC = 1024;   // embed
constexpr int kH = 4096;   // hidden
constexpr int kE = 8;      // experts
constexpr int kL = 1024;   // capacity = 2.0*T/E

constexpr int BM = 128, BN = 128, BK = 32;  // block tile
constexpr int LDT = 48;                     // padded LDS row stride (96B, 16B-aligned)
}

union Frag16 { v16bf v; v8bf h[2]; };

// CDNA5 async global->LDS staging (ASYNCcnt), guarded so a missing builtin
// falls back to synchronous vector copies.
#if defined(__AMDGCN__) && \
    __has_builtin(__builtin_amdgcn_global_load_async_to_lds_b128) && \
    __has_builtin(__builtin_amdgcn_s_wait_asynccnt)
#define MOE_ASYNC 1
#else
#define MOE_ASYNC 0
#endif

#if MOE_ASYNC
__device__ __forceinline__ void async_cp16(const __bf16* g, __bf16* l) {
  // Builtin signature (from compiler diagnostics): (v4i AS1*, v4i AS3*, Ii, Ii)
  __builtin_amdgcn_global_load_async_to_lds_b128(
      (__attribute__((address_space(1))) vi4*)g,
      (__attribute__((address_space(3))) vi4*)l, 0, 0);
}
#endif

// ---------------------------------------------------------------------------
// f32 -> bf16 elementwise convert (n multiple of 4)
// ---------------------------------------------------------------------------
__global__ void f32_to_bf16_kernel(const float* __restrict__ src,
                                   __bf16* __restrict__ dst, int n) {
  int i = (blockIdx.x * blockDim.x + threadIdx.x) * 4;
  if (i + 3 < n) {
    float4 v = *(const float4*)(src + i);
    dst[i + 0] = (__bf16)v.x;
    dst[i + 1] = (__bf16)v.y;
    dst[i + 2] = (__bf16)v.z;
    dst[i + 3] = (__bf16)v.w;
  }
}

// ---------------------------------------------------------------------------
// Fused f32->bf16 convert + transpose: src (slab, R, C) -> dst (slab, C, R).
// 32x32 LDS tiles, coalesced both directions.
// ---------------------------------------------------------------------------
__global__ __launch_bounds__(256)
void transpose_bf16_kernel(const float* __restrict__ src,
                           __bf16* __restrict__ dst, int R, int C) {
  __shared__ __bf16 tile[32][33];
  const int slab = blockIdx.z;
  const int r0 = blockIdx.y * 32;
  const int c0 = blockIdx.x * 32;
  const float* s = src + (size_t)slab * R * C;
  __bf16* d = dst + (size_t)slab * R * C;
  const int tid = threadIdx.x;
#pragma unroll
  for (int i = 0; i < 4; ++i) {
    const int idx = tid + i * 256;
    const int r = idx >> 5, c = idx & 31;
    tile[r][c] = (__bf16)s[(size_t)(r0 + r) * C + c0 + c];
  }
  __syncthreads();
#pragma unroll
  for (int i = 0; i < 4; ++i) {
    const int idx = tid + i * 256;
    const int cc = idx >> 5, rr = idx & 31;
    d[(size_t)(c0 + cc) * R + r0 + rr] = tile[rr][cc];
  }
}

// ---------------------------------------------------------------------------
// Gate GEMV + softmax over E=8. One wave per token (wave32 reduction).
// probs layout: (B, E, T) = probs[(b*E+e)*T + t]
// ---------------------------------------------------------------------------
__global__ __launch_bounds__(256)
void gate_softmax_kernel(const float* __restrict__ x, const float* __restrict__ gw,
                         float* __restrict__ probs) {
  const int token = blockIdx.x * 8 + (threadIdx.x >> 5);   // flat b*T+t
  const int lane  = threadIdx.x & 31;
  const float* xp = x + (size_t)token * kC;
  float s[kE];
#pragma unroll
  for (int e = 0; e < kE; ++e) s[e] = 0.f;
  for (int k = lane; k < kC; k += 32) {
    float xv = xp[k];
    const float* g = gw + (size_t)k * kE;
#pragma unroll
    for (int e = 0; e < kE; ++e) s[e] += xv * g[e];
  }
#pragma unroll
  for (int off = 16; off > 0; off >>= 1) {
#pragma unroll
    for (int e = 0; e < kE; ++e) s[e] += __shfl_xor(s[e], off);
  }
  if (lane == 0) {
    float m = s[0];
#pragma unroll
    for (int e = 1; e < kE; ++e) m = fmaxf(m, s[e]);
    float p[kE], sum = 0.f;
#pragma unroll
    for (int e = 0; e < kE; ++e) { p[e] = expf(s[e] - m); sum += p[e]; }
    const float inv = 1.f / sum;
    const int b = token >> 12;             // T = 4096
    const int t = token & (kT - 1);
#pragma unroll
    for (int e = 0; e < kE; ++e)
      probs[((size_t)(b * kE + e)) * kT + t] = p[e] * inv;
  }
}

// ---------------------------------------------------------------------------
// Per-(b,e) top-k via full bitonic sort of 4096 (prob, idx) pairs in LDS.
// Descending by prob, ties -> lower index (matches jax.lax.top_k).
// ---------------------------------------------------------------------------
__global__ __launch_bounds__(1024)
void topk_sort_kernel(const float* __restrict__ probs,
                      float* __restrict__ topp, int* __restrict__ topidx) {
  __shared__ float sp[kT];
  __shared__ int   si[kT];
  const int be  = blockIdx.x;
  const int tid = threadIdx.x;
  for (int i = tid; i < kT; i += 1024) {
    sp[i] = probs[(size_t)be * kT + i];
    si[i] = i;
  }
  __syncthreads();
  for (int k = 2; k <= kT; k <<= 1) {
    for (int j = k >> 1; j > 0; j >>= 1) {
      for (int i = tid; i < kT; i += 1024) {
        const int ixj = i ^ j;
        if (ixj > i) {
          const bool desc = ((i & k) == 0);
          const float pa = sp[i], pb = sp[ixj];
          const int   ia = si[i], ib = si[ixj];
          const bool aBeforeB = (pa > pb) || (pa == pb && ia < ib);
          if (desc != aBeforeB) {
            sp[i] = pb; sp[ixj] = pa;
            si[i] = ib; si[ixj] = ia;
          }
        }
      }
      __syncthreads();
    }
  }
  if (tid < kL) {
    topp[be * kL + tid]   = sp[tid];
    topidx[be * kL + tid] = si[tid];
  }
}

// ---------------------------------------------------------------------------
// inv[(b*E+e)*T + token] = slot in that expert's list, else -1 (pre-memset).
// ---------------------------------------------------------------------------
__global__ void build_inverse_kernel(const int* __restrict__ topidx,
                                     int* __restrict__ inv) {
  const int be = blockIdx.x;
  const int i  = threadIdx.x;
  inv[(size_t)be * kT + topidx[be * kL + i]] = i;
}

// ---------------------------------------------------------------------------
// WMMA fragment load. 16-bit A 16x32 layout (ISA 7.12.2):
//   lanes 0-15:  M=lane,    K = {0..7, 16..23}        (kh=0)
//   lanes 16-31: M=lane-16, K = {8..15, 24..31}       (kh=1)
// B (32x16) mirrors this with N in lanes; both stored K-contiguous in LDS,
// so each fragment is two 16B ds_load_b128.
// ---------------------------------------------------------------------------
__device__ __forceinline__ v16bf load_frag(const __bf16* rowbase, int kh) {
  Frag16 f;
  f.h[0] = *(const v8bf*)(rowbase + kh * 8);
  f.h[1] = *(const v8bf*)(rowbase + 16 + kh * 8);
  return f.v;
}

// ---------------------------------------------------------------------------
// Unified MoE GEMM: out = op( A @ Bt^T )
//  GATHER=1: A rows gathered through top_idx (GEMM1), GELU epilogue -> bf16
//  GATHER=0: A = h slab (GEMM2), top_p row-scale epilogue -> f32
// Bt is pre-transposed weights, (E, N, K) row-major -> K-contiguous rows.
// 128x128 block tile, 8 waves of 32x64 (2x4 WMMA tiles), K-step 32,
// double-buffered LDS; async global->LDS staging when available.
// ---------------------------------------------------------------------------
template <bool GATHER, bool GELU>
__global__ __launch_bounds__(256)
void moe_gemm_kernel(const __bf16* __restrict__ Abase0,
                     const __bf16* __restrict__ Bt,
                     const int* __restrict__ topidx,
                     const float* __restrict__ topp,
                     __bf16* __restrict__ outb,
                     float* __restrict__ outf,
                     const int Kd, const int Nd) {
  __shared__ __align__(16) __bf16 As[2][BM][LDT];
  __shared__ __align__(16) __bf16 Bs[2][BN][LDT];
  __shared__ int sIdx[BM];

  const int tid = threadIdx.x;
  const int be  = blockIdx.z;
  const int e   = be & (kE - 1);
  const int m0  = blockIdx.y * BM;
  const int n0  = blockIdx.x * BN;

  const int wave = tid >> 5, lane = tid & 31;
  const int wm = wave >> 1, wn = wave & 1;     // 4x2 wave grid -> 32x64 per wave
  const int lr = lane & 15, kh = lane >> 4;

  const __bf16* Abase = GATHER ? Abase0 + (size_t)(be >> 3) * kT * Kd
                               : Abase0 + (size_t)be * kL * Kd;
  const __bf16* Bbase = Bt + (size_t)e * Nd * Kd;

  if constexpr (GATHER) {
    if (tid < BM) sIdx[tid] = topidx[be * kL + m0 + tid];
    __syncthreads();
  }

  // Stage one 128x32 A tile and one 128x32 B tile (both K-contiguous) into
  // LDS buffer `buf`. 512 16B chunks per tile, 2 per thread.
  auto stage = [&](int kb, int buf) {
#pragma unroll
    for (int c = 0; c < 2; ++c) {
      const int chunk = tid + c * 256;
      const int row = chunk >> 2;        // 0..127
      const int seg = chunk & 3;         // 8 bf16 each
      size_t arow;
      if constexpr (GATHER) arow = (size_t)sIdx[row];
      else                  arow = (size_t)(m0 + row);
      const __bf16* ga = Abase + arow * Kd + kb + seg * 8;
      const __bf16* gb = Bbase + (size_t)(n0 + row) * Kd + kb + seg * 8;
#if MOE_ASYNC
      async_cp16(ga, &As[buf][row][seg * 8]);
      async_cp16(gb, &Bs[buf][row][seg * 8]);
#else
      *(v8bf*)&As[buf][row][seg * 8] = *(const v8bf*)ga;
      *(v8bf*)&Bs[buf][row][seg * 8] = *(const v8bf*)gb;
#endif
    }
  };

  v8f acc[2][4];
#pragma unroll
  for (int mt = 0; mt < 2; ++mt)
#pragma unroll
    for (int nt = 0; nt < 4; ++nt)
#pragma unroll
      for (int i = 0; i < 8; ++i) acc[mt][nt][i] = 0.f;

  stage(0, 0);

  int t = 0;
  for (int kb = 0; kb < Kd; kb += BK, t ^= 1) {
#if MOE_ASYNC
    __builtin_amdgcn_s_wait_asynccnt(0);
#endif
    __syncthreads();                       // tile t resident, buffer t^1 free
    if (kb + BK < Kd) stage(kb + BK, t ^ 1);

    v16bf afr[2], bfr[4];
#pragma unroll
    for (int mt = 0; mt < 2; ++mt)
      afr[mt] = load_frag(&As[t][wm * 32 + mt * 16 + lr][0], kh);
#pragma unroll
    for (int nt = 0; nt < 4; ++nt)
      bfr[nt] = load_frag(&Bs[t][wn * 64 + nt * 16 + lr][0], kh);

#pragma unroll
    for (int mt = 0; mt < 2; ++mt)
#pragma unroll
      for (int nt = 0; nt < 4; ++nt)
        acc[mt][nt] = __builtin_amdgcn_wmma_f32_16x16x32_bf16(
            false, afr[mt], false, bfr[nt], (short)0, acc[mt][nt], false, false);
  }

  // Epilogue. C/D layout: row = base + (lane>>4)*8 + i, col = base + (lane&15)
#pragma unroll
  for (int mt = 0; mt < 2; ++mt) {
#pragma unroll
    for (int nt = 0; nt < 4; ++nt) {
      const int colg = n0 + wn * 64 + nt * 16 + lr;
#pragma unroll
      for (int i = 0; i < 8; ++i) {
        const int rowg = m0 + wm * 32 + mt * 16 + kh * 8 + i;
        float v = acc[mt][nt][i];
        if constexpr (GELU) {
          v = 0.5f * v * (1.f + erff(v * 0.70710678118654752f));
          outb[((size_t)be * kL + rowg) * Nd + colg] = (__bf16)v;
        } else {
          outf[((size_t)be * kL + rowg) * Nd + colg] = v * topp[be * kL + rowg];
        }
      }
    }
  }
}

// ---------------------------------------------------------------------------
// Deterministic combine: out[b,t,c] = sum_e y[b,e,slot(b,e,t),c]
// ---------------------------------------------------------------------------
__global__ __launch_bounds__(256)
void combine_kernel(const float* __restrict__ y, const int* __restrict__ inv,
                    float* __restrict__ out) {
  const size_t idx = (size_t)blockIdx.x * 256 + threadIdx.x;
  const int c  = (int)(idx & (kC - 1));
  const int bt = (int)(idx >> 10);          // C = 1024
  const int t  = bt & (kT - 1);
  const int b  = bt >> 12;                  // T = 4096
  float acc = 0.f;
#pragma unroll
  for (int e = 0; e < kE; ++e) {
    const int s = inv[(size_t)(b * kE + e) * kT + t];
    if (s >= 0)
      acc += y[((size_t)(b * kE + e) * kL + s) * kC + c];
  }
  out[idx] = acc;
}

// ---------------------------------------------------------------------------
// Launch
// ---------------------------------------------------------------------------
extern "C" void kernel_launch(void* const* d_in, const int* in_sizes, int n_in,
                              void* d_out, int out_size, void* d_ws, size_t ws_size,
                              hipStream_t stream) {
  (void)in_sizes; (void)n_in; (void)out_size; (void)ws_size;

  const float* x  = (const float*)d_in[0];   // (B,T,C)
  const float* gw = (const float*)d_in[1];   // (C,E)
  const float* w1 = (const float*)d_in[2];   // (E,C,H)
  const float* w2 = (const float*)d_in[3];   // (E,H,C)
  float* out = (float*)d_out;                // (B,T,C)

  char* ws = (char*)d_ws;
  size_t off = 0;
  auto alloc = [&](size_t bytes) -> char* {
    char* p = ws + off;
    off += (bytes + 255) & ~(size_t)255;
    return p;
  };

  __bf16* xb    = (__bf16*)alloc((size_t)kB * kT * kC * 2);        //  32 MB
  __bf16* w1t   = (__bf16*)alloc((size_t)kE * kC * kH * 2);        //  64 MB (E,H,C)
  __bf16* w2t   = (__bf16*)alloc((size_t)kE * kH * kC * 2);        //  64 MB (E,C,H)
  __bf16* hbuf  = (__bf16*)alloc((size_t)kB * kE * kL * kH * 2);   // 256 MB
  float*  ybuf  = (float*) alloc((size_t)kB * kE * kL * kC * 4);   // 128 MB
  float*  probs = (float*) alloc((size_t)kB * kE * kT * 4);
  float*  topp  = (float*) alloc((size_t)kB * kE * kL * 4);
  int*    topix = (int*)   alloc((size_t)kB * kE * kL * 4);
  int*    inv   = (int*)   alloc((size_t)kB * kE * kT * 4);

  const int nx = kB * kT * kC;       // 16,777,216

  f32_to_bf16_kernel<<<nx / 1024, 256, 0, stream>>>(x, xb, nx);
  // w1 (E,C,H) -> w1t (E,H,C);  w2 (E,H,C) -> w2t (E,C,H)
  transpose_bf16_kernel<<<dim3(kH / 32, kC / 32, kE), 256, 0, stream>>>(w1, w1t, kC, kH);
  transpose_bf16_kernel<<<dim3(kC / 32, kH / 32, kE), 256, 0, stream>>>(w2, w2t, kH, kC);

  gate_softmax_kernel<<<kB * kT / 8, 256, 0, stream>>>(x, gw, probs);

  (void)hipMemsetAsync(inv, 0xFF, (size_t)kB * kE * kT * 4, stream);

  topk_sort_kernel<<<kB * kE, 1024, 0, stream>>>(probs, topp, topix);
  build_inverse_kernel<<<kB * kE, kL, 0, stream>>>(topix, inv);

  // GEMM1: h = gelu(gather(x) @ w1), bf16
  moe_gemm_kernel<true, true><<<dim3(kH / BN, kL / BM, kB * kE), 256, 0, stream>>>(
      xb, w1t, topix, nullptr, hbuf, nullptr, kC, kH);
  // GEMM2: y = (h @ w2) * top_p, f32
  moe_gemm_kernel<false, false><<<dim3(kC / BN, kL / BM, kB * kE), 256, 0, stream>>>(
      hbuf, w2t, nullptr, topp, nullptr, ybuf, kH, kC);

  combine_kernel<<<(unsigned)((size_t)kB * kT * kC / 256), 256, 0, stream>>>(
      ybuf, inv, out);
}
